// AutoEncoder_63625645523321
// MI455X (gfx1250) — compile-verified, matrix operand order checked by reading, not run on hardware
//
#include <hip/hip_runtime.h>
#include <hip/hip_bf16.h>
#include <cstdint>
#include <cstddef>

// ---------------------------------------------------------------------------
// CDNA5 (gfx1250) persistent LSTM seq2seq using V_WMMA_F32_16X16X32_BF16.
// Batch rows are independent through the recurrence -> one workgroup owns a
// 16-row tile for all 1023 timesteps; h/c live in LDS, weights are packed
// bf16 fragments resident in L2 and re-streamed every step. An opaque zero
// *offset* (not pointer) defeats LICM while keeping loads on the GLOBAL path,
// and the K-loop is software double-buffered to hide L2 latency under WMMA.
// ---------------------------------------------------------------------------

typedef __attribute__((ext_vector_type(16))) __bf16 v16bf;
typedef __attribute__((ext_vector_type(8)))  float  v8f;

#define B_      512
#define T_      512
#define D_      64
#define H_      256
#define G_      1024     // 4*H (gate columns)
#define KC_     320      // D + H  (combined K)
#define KT_     10       // KC_/32 K-tiles for gate GEMM
#define NT_     64       // G_/16 N-tiles total
#define ROWS    16       // batch rows per workgroup (one WMMA M-tile)
#define MAXLEN_ 512
#define OKT_    8        // H_/32 K-tiles for out projection
#define ONT_    4        // D_/16 N-tiles for out projection

#define ENC_N   (NT_ * KT_ * 512)     // bf16 elems for one packed gate-weight
#define OUTW_N  (ONT_ * OKT_ * 512)   // bf16 elems for packed out_W

// ---------------------------------------------------------------------------
// helpers
// ---------------------------------------------------------------------------
__device__ __forceinline__ unsigned short f2bf(float f) {
  union { float f; unsigned u; } x; x.f = f;
  unsigned r = x.u + 0x7FFFu + ((x.u >> 16) & 1u);   // round-to-nearest-even
  return (unsigned short)(r >> 16);
}

__device__ __forceinline__ float sigf(float x) {
  return 1.0f / (1.0f + __expf(-x));
}

// A-matrix 16x32 bf16 fragment swizzle (ISA 7.12.2):
//   lane = (row&15) | (((k&8)>>3)<<4) ; slot = (k&7) | (((k>>4)&1)<<3)
__device__ __forceinline__ int aswz(int r, int k) {
  int kt = k >> 5, kk = k & 31;
  int lane = (r & 15) | (((kk >> 3) & 1) << 4);
  int pos  = (kk & 7) | (((kk >> 4) & 1) << 3);
  return (kt * 32 + lane) * 16 + pos;
}

// ---------------------------------------------------------------------------
// one-time weight repack: fp32 -> bf16 in WMMA B-fragment order
//   fragment(nt, kt) = 512 contiguous bf16 = [lane 0..31][slot 0..15]
//   (k, n): lane = (n&15) | (((k&8)>>3)<<4), slot = (k&7) | (((k>>4)&1)<<3)
// ---------------------------------------------------------------------------
__global__ void __launch_bounds__(256)
pack_weights(const float* __restrict__ eWih, const float* __restrict__ eWhh,
             const float* __restrict__ eBih, const float* __restrict__ eBhh,
             const float* __restrict__ dWih, const float* __restrict__ dWhh,
             const float* __restrict__ dBih, const float* __restrict__ dBhh,
             const float* __restrict__ oW,   const float* __restrict__ oB,
             unsigned short* __restrict__ wsU, float* __restrict__ wsF)
{
  int idx = blockIdx.x * 256 + threadIdx.x;
  if (idx < 2 * ENC_N) {
    int which = (idx >= ENC_N) ? 1 : 0;
    int o   = which ? (idx - ENC_N) : idx;
    int nt  = o / (KT_ * 512);
    int rem = o % (KT_ * 512);
    int kt  = rem / 512;
    int q   = rem % 512;
    int lane = q >> 4, pos = q & 15;
    int n  = nt * 16 + (lane & 15);
    int kk = (pos & 7) | ((lane >> 4) << 3) | (((pos >> 3) & 1) << 4);
    int k  = kt * 32 + kk;
    const float* Wih = which ? dWih : eWih;
    const float* Whh = which ? dWhh : eWhh;
    float v = (k < D_) ? Wih[n * D_ + k] : Whh[n * H_ + (k - D_)];
    wsU[idx] = f2bf(v);
  } else if (idx < 2 * ENC_N + OUTW_N) {
    int o   = idx - 2 * ENC_N;
    int nt  = o / (OKT_ * 512);
    int rem = o % (OKT_ * 512);
    int kt  = rem / 512;
    int q   = rem % 512;
    int lane = q >> 4, pos = q & 15;
    int n  = nt * 16 + (lane & 15);
    int kk = (pos & 7) | ((lane >> 4) << 3) | (((pos >> 3) & 1) << 4);
    int k  = kt * 32 + kk;
    wsU[idx] = f2bf(oW[n * H_ + k]);              // B = out_W^T (256x64)
  } else {
    int o = idx - (2 * ENC_N + OUTW_N);
    if (o < G_)                wsF[o] = eBih[o] + eBhh[o];
    else if (o < 2 * G_)       wsF[o] = dBih[o - G_] + dBhh[o - G_];
    else if (o < 2 * G_ + D_)  wsF[o] = oB[o - 2 * G_];
  }
}

// ---------------------------------------------------------------------------
// per-step gate GEMM: gates(16 x 1024) = Xcat(16 x 320) @ Wcat^T
// Each wave owns 128 gate columns: 10 K-tiles x 8 N-tiles of
// v_wmma_f32_16x16x32_bf16, fp32 accumulation. Double-buffered over kt.
// ---------------------------------------------------------------------------
__device__ __forceinline__ void gemm_gates(const unsigned short* __restrict__ W,
                                           const v16bf* __restrict__ Af,
                                           int lane, int wave,
                                           float (*gatesS)[G_ + 4])
{
  // Opaque zero offset: executed every call, so weight-fragment loads cannot
  // be hoisted out of the time loop; base pointer stays provably GLOBAL.
  unsigned off = 0;
  asm volatile("" : "+s"(off));
  const v16bf* Bf = (const v16bf*)(W + off);   // fragment f at Bf[f*32 + lane]

  v8f acc[8] = {};

  // prologue: K-tile 0
  v16bf a = Af[lane];
  v16bf b[8];
#pragma unroll
  for (int nt = 0; nt < 8; ++nt)
    b[nt] = Bf[(size_t)((wave * 8 + nt) * KT_) * 32 + lane];

#pragma unroll
  for (int kt = 0; kt < KT_; ++kt) {
    v16bf an;
    v16bf bn[8];
    if (kt + 1 < KT_) {                        // prefetch next K-tile
      an = Af[(kt + 1) * 32 + lane];
#pragma unroll
      for (int nt = 0; nt < 8; ++nt)
        bn[nt] = Bf[(size_t)((wave * 8 + nt) * KT_ + kt + 1) * 32 + lane];
    }
#pragma unroll
    for (int nt = 0; nt < 8; ++nt)
      acc[nt] = __builtin_amdgcn_wmma_f32_16x16x32_bf16(
          false, a, false, b[nt], (short)0, acc[nt], false, false);
    if (kt + 1 < KT_) {
      a = an;
#pragma unroll
      for (int nt = 0; nt < 8; ++nt) b[nt] = bn[nt];
    }
  }

  // scatter C fragments: VGPR v holds M=v (lanes 0-15) / M=v+8 (lanes 16-31)
#pragma unroll
  for (int nt = 0; nt < 8; ++nt) {
    int n0 = wave * 128 + nt * 16 + (lane & 15);
    int r0 = (lane >> 4) * 8;
#pragma unroll
    for (int v = 0; v < 8; ++v)
      gatesS[r0 + v][n0] = acc[nt][v];
  }
}

// ---------------------------------------------------------------------------
// persistent seq2seq kernel: 32 WGs x 256 threads (8 waves)
// ---------------------------------------------------------------------------
__global__ void __launch_bounds__(256)
lstm_seq2seq(const float* __restrict__ xin,   // (B,T,D)
             const float* __restrict__ tgt,   // (B,MAXLEN,D)
             const int*   __restrict__ lens,  // (B,)
             const unsigned short* __restrict__ encW,
             const unsigned short* __restrict__ decW,
             const unsigned short* __restrict__ outW,
             const float* __restrict__ biasE,
             const float* __restrict__ biasD,
             const float* __restrict__ outBp,
             float* __restrict__ out)         // (B,MAXLEN,D)
{
  __shared__ alignas(64) unsigned short XcatA[KT_ * 32 * 16]; // swizzled [x|h] bf16
  __shared__ float gatesS[ROWS][G_ + 4];
  __shared__ float cS[ROWS][H_];
  __shared__ float yS[ROWS][D_];
  __shared__ float biasS[G_];
  __shared__ float outbS[D_];
  __shared__ int   lenS[ROWS];

  const int tid  = threadIdx.x;
  const int wave = tid >> 5;
  const int lane = tid & 31;
  const int rb   = blockIdx.x * ROWS;

  // init: h = 0 (whole Xcat buffer), c = 0, enc biases, lens, output t=0 slice
  for (int i = tid; i < KT_ * 32 * 16; i += 256) XcatA[i] = 0;
  for (int i = tid; i < ROWS * H_; i += 256) (&cS[0][0])[i] = 0.0f;
  for (int i = tid; i < G_; i += 256) biasS[i] = biasE[i];
  if (tid < D_)   outbS[tid] = outBp[tid];
  if (tid < ROWS) lenS[tid]  = lens[rb + tid];
  for (int i = tid; i < ROWS * D_; i += 256) {
    int r = i >> 6, d = i & 63;
    out[((size_t)(rb + r) * MAXLEN_ + 0) * D_ + d] = 0.0f;
  }
  __syncthreads();

  const v16bf* Af = (const v16bf*)XcatA;

  // ------------------------------ encoder ------------------------------
  for (int t = 0; t < T_; ++t) {
    // stage x_t bf16 into swizzled A panel (k in [0,64))
    for (int i = tid; i < ROWS * D_; i += 256) {
      int r = i >> 6, d = i & 63;
      const float* p = &xin[((size_t)(rb + r) * T_ + t) * D_ + d];
      XcatA[aswz(r, d)] = f2bf(*p);
      if (t + 1 < T_) __builtin_prefetch(p + D_, 0, 1);   // global_prefetch_b8
    }
    __syncthreads();

    gemm_gates(encW, Af, lane, wave, gatesS);
    __syncthreads();

    // elementwise LSTM update; thread tid owns hidden unit j = tid
    {
      int j = tid;
      for (int r = 0; r < ROWS; ++r) {
        float iv = sigf (gatesS[r][j        ] + biasS[j        ]);
        float fv = sigf (gatesS[r][j + 256  ] + biasS[j + 256  ]);
        float gv = tanhf(gatesS[r][j + 512  ] + biasS[j + 512  ]);
        float ov = sigf (gatesS[r][j + 768  ] + biasS[j + 768  ]);
        float c2 = fv * cS[r][j] + iv * gv;
        float h2 = ov * tanhf(c2);
        if (t < lenS[r]) {                       // masked state update
          cS[r][j] = c2;
          XcatA[aswz(r, D_ + j)] = f2bf(h2);
        }
      }
    }
    __syncthreads();
  }

  // swap to decoder biases
  for (int i = tid; i < G_; i += 256) biasS[i] = biasD[i];
  __syncthreads();

  // ------------------------------ decoder ------------------------------
  for (int dt = 0; dt < MAXLEN_ - 1; ++dt) {
    // stage x (target[:,0,:] at dt==0, else previous y)
    for (int i = tid; i < ROWS * D_; i += 256) {
      int r = i >> 6, d = i & 63;
      float xv = (dt == 0) ? tgt[(size_t)(rb + r) * MAXLEN_ * D_ + d] : yS[r][d];
      XcatA[aswz(r, d)] = f2bf(xv);
    }
    __syncthreads();

    gemm_gates(decW, Af, lane, wave, gatesS);
    __syncthreads();

    {
      int j = tid;
      for (int r = 0; r < ROWS; ++r) {
        float iv = sigf (gatesS[r][j        ] + biasS[j        ]);
        float fv = sigf (gatesS[r][j + 256  ] + biasS[j + 256  ]);
        float gv = tanhf(gatesS[r][j + 512  ] + biasS[j + 512  ]);
        float ov = sigf (gatesS[r][j + 768  ] + biasS[j + 768  ]);
        float c2 = fv * cS[r][j] + iv * gv;
        float h2 = ov * tanhf(c2);
        cS[r][j] = c2;                            // no mask in decoder scan
        XcatA[aswz(r, D_ + j)] = f2bf(h2);
      }
    }
    __syncthreads();

    // out projection y = h @ out_W^T + out_b  (waves 0..3, h = Xcat K [64,320))
    if (wave < ONT_) {
      unsigned off = 0;
      asm volatile("" : "+s"(off));     // defeat LICM, keep GLOBAL addressing
      const v16bf* Of = (const v16bf*)(outW + off);
      v8f yacc = {};
#pragma unroll
      for (int kt = 0; kt < OKT_; ++kt) {
        v16bf a = Af[(kt + 2) * 32 + lane];
        v16bf b = Of[(size_t)(wave * OKT_ + kt) * 32 + lane];
        yacc = __builtin_amdgcn_wmma_f32_16x16x32_bf16(
            false, a, false, b, (short)0, yacc, false, false);
      }
      int n0 = wave * 16 + (lane & 15);
      int r0 = (lane >> 4) * 8;
#pragma unroll
      for (int v = 0; v < 8; ++v)
        yS[r0 + v][n0] = yacc[v] + outbS[n0];
    }
    __syncthreads();

    // write masked output slice t = dt+1
    int tt = dt + 1;
    for (int i = tid; i < ROWS * D_; i += 256) {
      int r = i >> 6, d = i & 63;
      out[((size_t)(rb + r) * MAXLEN_ + tt) * D_ + d] =
          (tt < lenS[r]) ? yS[r][d] : 0.0f;
    }
    __syncthreads();
  }
}

// ---------------------------------------------------------------------------
// host launch
// ---------------------------------------------------------------------------
extern "C" void kernel_launch(void* const* d_in, const int* in_sizes, int n_in,
                              void* d_out, int out_size, void* d_ws, size_t ws_size,
                              hipStream_t stream)
{
  (void)in_sizes; (void)n_in; (void)out_size; (void)ws_size;

  const float* xin  = (const float*)d_in[0];
  const float* tgt  = (const float*)d_in[1];
  const float* eWih = (const float*)d_in[2];
  const float* eWhh = (const float*)d_in[3];
  const float* eBih = (const float*)d_in[4];
  const float* eBhh = (const float*)d_in[5];
  const float* dWih = (const float*)d_in[6];
  const float* dWhh = (const float*)d_in[7];
  const float* dBih = (const float*)d_in[8];
  const float* dBhh = (const float*)d_in[9];
  const float* oW   = (const float*)d_in[10];
  const float* oB   = (const float*)d_in[11];
  const int*   lens = (const int*)d_in[12];

  unsigned short* wsU = (unsigned short*)d_ws;
  float* wsF = (float*)((char*)d_ws + (size_t)(2 * ENC_N + OUTW_N) * sizeof(unsigned short));

  int packElems  = 2 * ENC_N + OUTW_N + 2 * G_ + D_;
  int packBlocks = (packElems + 255) / 256;
  pack_weights<<<packBlocks, 256, 0, stream>>>(eWih, eWhh, eBih, eBhh,
                                               dWih, dWhh, dBih, dBhh,
                                               oW, oB, wsU, wsF);

  const unsigned short* encW = wsU;
  const unsigned short* decW = wsU + ENC_N;
  const unsigned short* outW = wsU + 2 * ENC_N;
  const float* biasE = wsF;
  const float* biasD = wsF + G_;
  const float* outBp = wsF + 2 * G_;

  lstm_seq2seq<<<B_ / ROWS, 256, 0, stream>>>(xin, tgt, lens, encW, decW, outW,
                                              biasE, biasD, outBp, (float*)d_out);
}
